// STBlock_54065048322482
// MI455X (gfx1250) — matrix operand order repaired and checked by reading, not compiled, not run on hardware
//
#include <hip/hip_runtime.h>
#include <hip/hip_bf16.h>

// ---------------------------------------------------------------------------
// ST-GCN block for MI455X (gfx1250, wave32, WMMA).
// All contractions run through v_wmma_f32_16x16x32_bf16 (bf16 operands,
// f32 accumulate). Strides are template constants so fragment addressing
// folds to immediate offsets; each wave computes a 32x32 C tile (4 WMMAs
// per 32-wide K step). Intermediates live in d_ws (~113 MB, L2-resident).
// ---------------------------------------------------------------------------

typedef __attribute__((ext_vector_type(16))) __bf16 v16bf;
typedef __attribute__((ext_vector_type(8)))  __bf16 bfx8;
typedef __attribute__((ext_vector_type(8)))  float  v8f;

// ---------------------------------------------------------------------------
// f32 -> bf16 conversion (grid-stride)
// ---------------------------------------------------------------------------
__global__ void cvt_f32_bf16(const float* __restrict__ s, __bf16* __restrict__ d, long n) {
  long i = (long)blockIdx.x * blockDim.x + threadIdx.x;
  long stride = (long)gridDim.x * blockDim.x;
  for (; i < n; i += stride) d[i] = (__bf16)s[i];
}

// ---------------------------------------------------------------------------
// Fragment loaders (CDNA5 ISA 7.12.2 layouts). AK/BK are compile-time.
// A 16x32: lane<16 holds K {0..7,16..23} of row m, lane>=16 {8..15,24..31}.
// B 32x16: lane<16 holds K {0..7,16..23} of col n, lane>=16 {8..15,24..31}.
// ---------------------------------------------------------------------------
template <long AK>
__device__ __forceinline__ v16bf load_a_frag(const __bf16* __restrict__ p, int k0, int hi) {
  v16bf a;
  if constexpr (AK == 1) {
    bfx8 lo  = *(const bfx8*)(p + (long)(k0 + hi * 8));
    bfx8 hi8 = *(const bfx8*)(p + (long)(k0 + 16 + hi * 8));
    #pragma unroll
    for (int j = 0; j < 8; ++j) { a[j] = lo[j]; a[8 + j] = hi8[j]; }
  } else {
    #pragma unroll
    for (int j = 0; j < 8; ++j) {
      a[j]     = p[(long)(k0 + hi * 8 + j)      * AK];
      a[8 + j] = p[(long)(k0 + 16 + hi * 8 + j) * AK];
    }
  }
  return a;
}

template <long BK>
__device__ __forceinline__ v16bf load_b_frag(const __bf16* __restrict__ p, int k0, int hi) {
  v16bf b;
  if constexpr (BK == 1) {
    bfx8 b0 = *(const bfx8*)(p + (long)(k0 + hi * 8));
    bfx8 b1 = *(const bfx8*)(p + (long)(k0 + 16 + hi * 8));
    #pragma unroll
    for (int j = 0; j < 8; ++j) { b[j] = b0[j]; b[8 + j] = b1[j]; }
  } else {
    #pragma unroll
    for (int j = 0; j < 16; ++j)
      b[j] = p[(long)(k0 + (j < 8 ? hi * 8 + j : 16 + hi * 8 + (j - 8))) * BK];
  }
  return b;
}

// ---------------------------------------------------------------------------
// Batched bf16 WMMA GEMM:  C = alpha * A(MxK) x B(KxN) + beta * D
// All strides compile-time. Block = 128 threads = 4 waves in a 2x2 grid;
// each wave owns a 32x32 tile (2 A-frags x 2 B-frags -> 4 WMMAs / K-step).
// Grid: (N/64, M/64, batch).
// ---------------------------------------------------------------------------
template <long AB, long AM, long AK,
          long BB, long BK, long BN,
          long DB, long DM, long DN,
          long CB, long CM, long CN, bool HASD>
__global__ __launch_bounds__(128) void gemm_tile(
    const __bf16* __restrict__ A, const __bf16* __restrict__ B,
    const __bf16* __restrict__ D, __bf16* __restrict__ C,
    int K, float alpha, float beta)
{
  const int lane = threadIdx.x & 31;
  const int wid  = threadIdx.x >> 5;
  const int wy   = wid >> 1, wx = wid & 1;
  const int mloc = lane & 15, hi = lane >> 4;
  const long bz  = blockIdx.z;
  const long m0  = (long)blockIdx.y * 64 + wy * 32;
  const long n0  = (long)blockIdx.x * 64 + wx * 32;

  const __bf16* A0 = A + bz * AB + (m0 + mloc) * AM;
  const __bf16* A1 = A0 + 16 * AM;
  const __bf16* B0 = B + bz * BB + (n0 + mloc) * BN;
  const __bf16* B1 = B0 + 16 * BN;

  v8f acc00 = {0.f,0.f,0.f,0.f,0.f,0.f,0.f,0.f};
  v8f acc01 = acc00, acc10 = acc00, acc11 = acc00;

  for (int k0 = 0; k0 < K; k0 += 32) {
    // prefetch next-next K tile into cache (global_prefetch_b8)
    if (k0 + 64 < K) {
      __builtin_prefetch((const void*)(A0 + (long)(k0 + 64) * AK), 0, 1);
      __builtin_prefetch((const void*)(B0 + (long)(k0 + 64) * BK), 0, 1);
    }
    v16bf a0 = load_a_frag<AK>(A0, k0, hi);
    v16bf a1 = load_a_frag<AK>(A1, k0, hi);
    v16bf b0 = load_b_frag<BK>(B0, k0, hi);
    v16bf b1 = load_b_frag<BK>(B1, k0, hi);
    acc00 = __builtin_amdgcn_wmma_f32_16x16x32_bf16(false, a0, false, b0, (short)0, acc00, false, false);
    acc01 = __builtin_amdgcn_wmma_f32_16x16x32_bf16(false, a0, false, b1, (short)0, acc01, false, false);
    acc10 = __builtin_amdgcn_wmma_f32_16x16x32_bf16(false, a1, false, b0, (short)0, acc10, false, false);
    acc11 = __builtin_amdgcn_wmma_f32_16x16x32_bf16(false, a1, false, b1, (short)0, acc11, false, false);
  }

  // epilogue: VGPR r -> row my*16 + hi*8 + r, col nx*16 + mloc
  v8f accs[2][2] = {{acc00, acc01}, {acc10, acc11}};
  #pragma unroll
  for (int my = 0; my < 2; ++my) {
    #pragma unroll
    for (int nx = 0; nx < 2; ++nx) {
      #pragma unroll
      for (int r = 0; r < 8; ++r) {
        long row = m0 + my * 16 + hi * 8 + r;
        long col = n0 + nx * 16 + mloc;
        float v = alpha * accs[my][nx][r];
        if constexpr (HASD) v += beta * (float)D[bz * DB + row * DM + col * DN];
        C[bz * CB + row * CM + col * CN] = (__bf16)v;
      }
    }
  }
}

template <long AB, long AM, long AK,
          long BB, long BK, long BN,
          long DB, long DM, long DN,
          long CB, long CM, long CN, bool HASD>
static void launch_gemm(const __bf16* A, const __bf16* B, const __bf16* D, __bf16* C,
                        int M, int N, int K, int batch,
                        float alpha, float beta, hipStream_t stream)
{
  dim3 grid((unsigned)(N / 64), (unsigned)(M / 64), (unsigned)batch);
  gemm_tile<AB, AM, AK, BB, BK, BN, DB, DM, DN, CB, CM, CN, HASD>
      <<<grid, 128, 0, stream>>>(A, B, D, C, K, alpha, beta);
}

// ---------------------------------------------------------------------------
// Attention scores: s[b,i,j] = sum_k sigmoid(lhs[b,i,k] + rhs[b,j,k]) + V[i,j]
// then softmax over j. One block per (b,i) row; L threads (L = 256 or 64).
// ---------------------------------------------------------------------------
template <int L>
__global__ void attn_score(const __bf16* __restrict__ lhs,
                           const __bf16* __restrict__ rhs,
                           const float* __restrict__ V,
                           __bf16* __restrict__ out)
{
  const int i = blockIdx.x;
  const int b = blockIdx.y;
  const int j = threadIdx.x;
  __shared__ float lrow[L];
  __shared__ float red[L];

  lrow[j] = (float)lhs[((long)b * L + i) * L + j];
  __syncthreads();

  const __bf16* rr = rhs + ((long)b * L + j) * L;
  float s = V[(long)i * L + j];
  #pragma unroll 4
  for (int k = 0; k < L; ++k) {
    float z = lrow[k] + (float)rr[k];
    s += 1.0f / (1.0f + __expf(-z));     // v_exp_f32
  }

  red[j] = s; __syncthreads();
  for (int off = L >> 1; off; off >>= 1) {
    if (j < off) red[j] = fmaxf(red[j], red[j + off]);
    __syncthreads();
  }
  float m = red[0]; __syncthreads();
  float e = __expf(s - m);
  red[j] = e; __syncthreads();
  for (int off = L >> 1; off; off >>= 1) {
    if (j < off) red[j] += red[j + off];
    __syncthreads();
  }
  float inv = 1.0f / red[0];
  out[((long)b * L + i) * L + j] = (__bf16)(e * inv);
}

// ---------------------------------------------------------------------------
// Fused epilogue per (b,n): (1,3) temporal conv + bias + ReLU, 1x1 residual
// conv + add + ReLU, write out[b,n,o,t].
// ---------------------------------------------------------------------------
__global__ __launch_bounds__(256) void fuse_out(
    const __bf16* __restrict__ g, const float* __restrict__ x,
    const float* __restrict__ tw, const float* __restrict__ tb,
    const float* __restrict__ rw, const float* __restrict__ rb,
    float* __restrict__ out)
{
  const int bn = blockIdx.x;                 // b*256 + n
  __shared__ float gs[64][65];               // [c][t]
  __shared__ float xs[64][65];
  const long base = (long)bn * 4096;

  for (int idx = threadIdx.x; idx < 4096; idx += 256) {
    gs[idx >> 6][idx & 63] = (float)g[base + idx];
    xs[idx >> 6][idx & 63] = x[base + idx];
  }
  __syncthreads();

  for (int idx = threadIdx.x; idx < 4096; idx += 256) {
    int o = idx >> 6, t = idx & 63;
    float acc  = tb[o];
    float racc = rb[o];
    const float* two = tw + (long)o * 64 * 3;
    const float* rwo = rw + (long)o * 64;
    #pragma unroll 4
    for (int c = 0; c < 64; ++c) {
      float g0 = (t > 0)  ? gs[c][t - 1] : 0.f;
      float g1 = gs[c][t];
      float g2 = (t < 63) ? gs[c][t + 1] : 0.f;
      acc  = fmaf(two[c * 3 + 0], g0, acc);
      acc  = fmaf(two[c * 3 + 1], g1, acc);
      acc  = fmaf(two[c * 3 + 2], g2, acc);
      racc = fmaf(rwo[c], xs[c][t], racc);
    }
    acc = fmaxf(acc, 0.f);
    out[base + idx] = fmaxf(acc + racc, 0.f);
  }
}

// ---------------------------------------------------------------------------
// Host orchestration
// ---------------------------------------------------------------------------
extern "C" void kernel_launch(void* const* d_in, const int* in_sizes, int n_in,
                              void* d_out, int out_size, void* d_ws, size_t ws_size,
                              hipStream_t stream)
{
  (void)in_sizes; (void)n_in; (void)out_size; (void)ws_size;
  const float* x    = (const float*)d_in[0];   // [8,256,64,64]
  const float* lap  = (const float*)d_in[1];   // [256,256]
  const float* W1s  = (const float*)d_in[2];   // [256,4096]
  const float* W2s  = (const float*)d_in[3];
  const float* Vs   = (const float*)d_in[4];   // [256,256]
  const float* W1t  = (const float*)d_in[5];   // [64,16384]
  const float* W2t  = (const float*)d_in[6];
  const float* Ve   = (const float*)d_in[7];   // [64,64]
  const float* cheb = (const float*)d_in[8];   // [3,64,64]
  const float* tw   = (const float*)d_in[9];   // [64,64,1,3]
  const float* tb   = (const float*)d_in[10];  // [64]
  const float* rw   = (const float*)d_in[11];  // [64,64,1,1]
  const float* rb   = (const float*)d_in[12];  // [64]
  float* out = (float*)d_out;                  // [8,256,64,64]

  // ---- workspace layout (bf16 elements), ~113 MB total -------------------
  __bf16* p       = (__bf16*)d_ws;
  __bf16* x_bf    = p;                 p += 8388608;   // [b,n,c,t]
  __bf16* W1s_bf  = p;                 p += 1048576;
  __bf16* W2s_bf  = p;                 p += 1048576;
  __bf16* W1t_bf  = p;                 p += 1048576;
  __bf16* W2t_bf  = p;                 p += 1048576;
  __bf16* lap_bf  = p;                 p += 65536;
  __bf16* cheb_bf = p;                 p += 12288;
  __bf16* lhs_s   = p;                 p += 524288;    // [b,256,256]
  __bf16* rhs_s   = p;                 p += 524288;
  __bf16* lhs_t   = p;                 p += 32768;     // [b,64,64]
  __bf16* rhs_t   = p;                 p += 32768;
  __bf16* S_bf    = p;                 p += 524288;    // [b,256,256]
  __bf16* E_bf    = p;                 p += 32768;     // [b,64,64]
  __bf16* xt_bf   = p;                 p += 8388608;   // x_time [b,(n,c),s]
  __bf16* terms   = p;                 p += 25165824;  // [b,n][k=3][c][s]
  __bf16* g_bf    = p;                 p += 8388608;   // [b,n,o,s]

  auto cvt = [&](const float* s, __bf16* d, long n) {
    long blk = (n + 255) / 256; if (blk > 4096) blk = 4096;
    cvt_f32_bf16<<<dim3((unsigned)blk), 256, 0, stream>>>(s, d, n);
  };

  // ---- stage bf16 operands ----------------------------------------------
  cvt(x,    x_bf,    8388608);
  cvt(W1s,  W1s_bf,  1048576);
  cvt(W2s,  W2s_bf,  1048576);
  cvt(W1t,  W1t_bf,  1048576);
  cvt(W2t,  W2t_bf,  1048576);
  cvt(lap,  lap_bf,  65536);
  cvt(cheb, cheb_bf, 12288);

  // ---- attention projections (WMMA) -------------------------------------
  // lhs_s[b,i,r] = sum_f x[b,i,f] * W1s[r,f]   (M=256,N=256,K=4096)
  launch_gemm<1048576,4096,1,  0,1,4096,  0,0,0,  65536,256,1, false>(
      x_bf, W1s_bf, nullptr, lhs_s, 256, 256, 4096, 8, 1.f, 0.f, stream);
  launch_gemm<1048576,4096,1,  0,1,4096,  0,0,0,  65536,256,1, false>(
      x_bf, W2s_bf, nullptr, rhs_s, 256, 256, 4096, 8, 1.f, 0.f, stream);
  // lhs_t[b,t,r]: A is x viewed [t, f=(n,c)] via strides (am=1, ak=64)
  launch_gemm<1048576,1,64,  0,1,16384,  0,0,0,  4096,64,1, false>(
      x_bf, W1t_bf, nullptr, lhs_t, 64, 64, 16384, 8, 1.f, 0.f, stream);
  launch_gemm<1048576,1,64,  0,1,16384,  0,0,0,  4096,64,1, false>(
      x_bf, W2t_bf, nullptr, rhs_t, 64, 64, 16384, 8, 1.f, 0.f, stream);

  // ---- sigmoid-sum + softmax --------------------------------------------
  attn_score<256><<<dim3(256, 8), 256, 0, stream>>>(lhs_s, rhs_s, Vs, S_bf);
  attn_score<64> <<<dim3(64, 8),  64,  0, stream>>>(lhs_t, rhs_t, Ve, E_bf);

  // ---- x_time[b,(n,c),s] = sum_t x[b,(n,c),t] * E[b,s,t] -----------------
  launch_gemm<1048576,64,1,  4096,1,64,  0,0,0,  1048576,64,1, false>(
      x_bf, E_bf, nullptr, xt_bf, 16384, 64, 64, 8, 1.f, 0.f, stream);

  // ---- Chebyshev terms, interleaved [b,n][k][c][s] -----------------------
  // term0 = S @ x_time   (M=256, N=4096, K=256)
  launch_gemm<65536,256,1,  1048576,4096,1,  0,0,0,  3145728,12288,1, false>(
      S_bf, xt_bf, nullptr, terms, 256, 4096, 256, 8, 1.f, 0.f, stream);
  // term1 = lap @ term0
  launch_gemm<0,256,1,  3145728,12288,1,  0,0,0,  3145728,12288,1, false>(
      lap_bf, terms, nullptr, terms + 4096, 256, 4096, 256, 8, 1.f, 0.f, stream);
  // term2 = 2 * lap @ term1 - term0
  launch_gemm<0,256,1,  3145728,12288,1,  3145728,12288,1,  3145728,12288,1, true>(
      lap_bf, terms + 4096, terms, terms + 8192, 256, 4096, 256, 8, 2.f, -1.f, stream);

  // ---- g[b,n,o,s] = sum_{k,i} cheb[k,i,o] * terms[b,n,k,i,s]  (K=192) ----
  launch_gemm<0,1,64,  12288,64,1,  0,0,0,  4096,64,1, false>(
      cheb_bf, terms, nullptr, g_bf, 64, 64, 192, 2048, 1.f, 0.f, stream);

  // ---- fused temporal conv + residual + ReLUs ----------------------------
  fuse_out<<<dim3(2048), 256, 0, stream>>>(g_bf, x, tw, tb, rw, rb, out);
}
